// HybridBinaryClassifier_65481071397668
// MI455X (gfx1250) — matrix operand order repaired and checked by reading, not compiled, not run on hardware
//
#include <hip/hip_runtime.h>
#include <math.h>

// ---------------------------------------------------------------------------
// Fused HybridBinaryClassifier inference for MI455X (gfx1250, wave32, WMMA).
// One block = 16 images end-to-end; activations staged in zero-padded f16 LDS
// tiles; all dense math on v_wmma_f32_16x16x32_f16; branch-free inner loops.
// ---------------------------------------------------------------------------

typedef __attribute__((ext_vector_type(16))) _Float16 v16h;
typedef __attribute__((ext_vector_type(8)))  _Float16 h8;
typedef __attribute__((ext_vector_type(4)))  _Float16 h4;
typedef __attribute__((ext_vector_type(8)))  float    v8f;

union V16H { v16h v; _Float16 h[16]; h8 q[2]; };
union V8F  { v8f  v; float    f[8];  };

__device__ __forceinline__ v8f wmma32(v16h a, v16h b, v8f c) {
    // (neg_a, A, neg_b, B, c_mod, C, reuse_a, reuse_b)
    return __builtin_amdgcn_wmma_f32_16x16x32_f16(false, a, false, b,
                                                  (short)0, c, false, false);
}

// A-matrix (16x32 f16) per-lane K mapping (ISA 7.12.2):
// lanes 0-15: elem e -> K = {0..7, 16..23}; lanes 16-31: K = {8..15, 24..31}
__device__ __forceinline__ int kofA(int e, int hi) {
    return ((e & 8) << 1) + (e & 7) + (hi ? 8 : 0);
}

#define B_TOTAL   8192
#define IPB       16
#define NBLK      (B_TOTAL / IPB)

// ---- LDS layout (units: halves). Overlays use disjoint lifetimes. ----------
// XP   : padded x     [16][3][34 rows x 36 stride], halo so conv1 needs no guards
#define XP_OFF      0
#define XP_IMG      3672            // 3*1224
#define XP_IC       1224            // 34*36
#define XP_ROW      36
#define XP_SZ       58752
// H1PRE: conv1 out    [16][6][15][15]
#define H1PRE_OFF   58752
#define H1PRE_SZ    21600
// H1P  : pooled h1    [16][6][16][16] padded (overlays XP after conv1)
#define H1P_OFF     0
#define H1P_IMG     1536
#define H1P_IC      256
#define H1P_SZ      24576
// H2PRE: conv2 out    [16][15][7][7]  (overlays H1PRE)
#define H2PRE_OFF   58752
// F    : features     [16][544] zero-padded K (after H1P)
#define F_OFF       24576
#define F_STR       544
// F1   : fc1 out      [16][128] zero-padded K
#define F1_OFF      33280
#define F1_STR      128
// F2   : fc2 out      [16][96]  zero-padded cols
#define F2_OFF      35328
#define F2_STR      96
#define SMEM_HALFS  80352
#define SMEM_BYTES  (SMEM_HALFS * 2)      // 160704 B; 2 blocks/WGP (320KB)

__global__ __launch_bounds__(256)
void hybrid_net_kernel(const float* __restrict__ x,
                       const float* __restrict__ w_conv1,
                       const float* __restrict__ g1, const float* __restrict__ b1,
                       const float* __restrict__ w_res,
                       const float* __restrict__ g2, const float* __restrict__ b2,
                       const float* __restrict__ w_sc,
                       const float* __restrict__ g3, const float* __restrict__ b3,
                       const float* __restrict__ w_fc1, const float* __restrict__ b_fc1,
                       const float* __restrict__ w_fc2, const float* __restrict__ b_fc2,
                       const float* __restrict__ w_fc3, const float* __restrict__ b_fc3,
                       float* __restrict__ out)
{
    extern __shared__ _Float16 smem[];
    const int tid  = threadIdx.x;
    const int lane = tid & 31;
    const int wave = tid >> 5;
    const int hi   = lane >> 4;
    const int lrow = lane & 15;        // A-row / B,D-column / image index
    const int g    = blockIdx.x;
    const float rs = rsqrtf(1.0f + 1e-5f);   // BN: g / sqrt(1+eps)

    // ------------- stage 0a: zero the padded x tile (halo) -----------------
    {
        uint4 z; z.x = 0u; z.y = 0u; z.z = 0u; z.w = 0u;
        uint4* p = (uint4*)smem;                   // 8 halves per store
        for (int i = tid; i < XP_SZ / 8; i += 256) p[i] = z;
    }
    __syncthreads();

    // ------------- stage 0b: x f32 -> f16 interior fill ---------------------
    {
        const float4* xg = (const float4*)(x + (size_t)g * IPB * 3072);
        for (int i = tid; i < IPB * 3072 / 4; i += 256) {
            float4 v = xg[i];
            int t   = i * 4;
            int img = t / 3072;  int r = t - img * 3072;
            int ic  = r >> 10;   r &= 1023;
            int yy  = r >> 5;    int xc = r & 31;
            h4 pk; pk.x = (_Float16)v.x; pk.y = (_Float16)v.y;
                   pk.z = (_Float16)v.z; pk.w = (_Float16)v.w;
            // row offset +1 (halo), col offset +4 (halo, keeps 8B alignment)
            *(h4*)&smem[XP_OFF + img*XP_IMG + ic*XP_IC + (yy+1)*XP_ROW + xc + 4] = pk;
        }
    }
    __syncthreads();

    // ------------- stage 1: conv1 5x5 s2 p1 + BN + ReLU ---------------------
    // GEMM: M = oc(6->16), N = img(16), K = 75->96. Branch-free B gathers.
    {
        V16H a1[3];
        int  offs1[3][16];                 // lane-constant im2col LDS offsets
        #pragma unroll
        for (int c = 0; c < 3; ++c) {
            #pragma unroll
            for (int e = 0; e < 16; ++e) {
                int ka = c*32 + kofA(e, hi);
                float w = (lrow < 6 && ka < 75) ? w_conv1[lrow*75 + ka] : 0.0f;
                a1[c].h[e] = (_Float16)w;
                int k  = c*32 + kofA(e, hi);
                int kk = (k < 75) ? k : 74;            // A has zeros there anyway
                int ic = kk/25, r = kk - ic*25, ky = r/5, kx = r - ky*5;
                offs1[c][e] = ic*XP_IC + ky*XP_ROW + kx;
            }
        }
        for (int p = wave; p < 225; p += 8) {          // wave-uniform
            int oy = p / 15, ox = p - oy * 15;
            // padded coords: row = 2oy+ky (halo +1 folded), col = 2ox+kx+3
            int pbase = XP_OFF + lrow*XP_IMG + oy*(2*XP_ROW) + ox*2 + 3;
            V8F acc;
            #pragma unroll
            for (int j = 0; j < 8; ++j) acc.f[j] = 0.0f;
            #pragma unroll
            for (int c = 0; c < 3; ++c) {
                V16H bm;
                #pragma unroll
                for (int e = 0; e < 16; ++e)
                    bm.h[e] = smem[pbase + offs1[c][e]];
                acc.v = wmma32(a1[c].v, bm.v, acc.v);
            }
            #pragma unroll
            for (int j = 0; j < 8; ++j) {
                int oc = hi * 8 + j;
                if (oc < 6) {
                    float y = fmaxf(acc.f[j] * (g1[oc]*rs) + b1[oc], 0.0f);
                    smem[H1PRE_OFF + lrow*1350 + oc*225 + oy*15 + ox] = (_Float16)y;
                }
            }
        }
    }
    __syncthreads();

    // ------------- stage 2a: zero padded h1 tile ----------------------------
    {
        uint4 z; z.x = 0u; z.y = 0u; z.z = 0u; z.w = 0u;
        uint4* p = (uint4*)smem;                       // H1P_OFF == 0
        for (int i = tid; i < H1P_SZ / 8; i += 256) p[i] = z;
    }
    __syncthreads();

    // ------------- stage 2b: maxpool 2x2 s1 -> h1 interior ------------------
    for (int idx = tid; idx < IPB * 6 * 14 * 14; idx += 256) {
        int img = idx / 1176; int r = idx - img * 1176;
        int oc  = r / 196;    r -= oc * 196;
        int y   = r / 14;     int xx = r - y * 14;
        const _Float16* s = &smem[H1PRE_OFF + img*1350 + oc*225 + y*15 + xx];
        float m = fmaxf(fmaxf((float)s[0], (float)s[1]),
                        fmaxf((float)s[15], (float)s[16]));
        smem[H1P_OFF + img*H1P_IMG + oc*H1P_IC + (y+1)*16 + (xx+1)] = (_Float16)m;
    }
    __syncthreads();

    // ------------- stage 3: conv_res 3x3 s2 p1 + conv_sc 1x1 s2, BN, add ----
    // GEMM: M = oc(15->16), N = img(16), K = 54->64 and 6->32.
    {
        V16H a2[2], asc;
        int  offs2[2][16], offsc[16];
        #pragma unroll
        for (int c = 0; c < 2; ++c) {
            #pragma unroll
            for (int e = 0; e < 16; ++e) {
                int ka = c*32 + kofA(e, hi);
                float w = (lrow < 15 && ka < 54) ? w_res[lrow*54 + ka] : 0.0f;
                a2[c].h[e] = (_Float16)w;
                int k  = c*32 + kofA(e, hi);
                int kk = (k < 54) ? k : 53;
                int ic = kk/9, r = kk - ic*9, ky = r/3, kx = r - ky*3;
                offs2[c][e] = ic*H1P_IC + ky*16 + kx;
            }
        }
        #pragma unroll
        for (int e = 0; e < 16; ++e) {
            int ka = kofA(e, hi);
            float w = (lrow < 15 && ka < 6) ? w_sc[lrow*6 + ka] : 0.0f;
            asc.h[e] = (_Float16)w;
            int k  = kofA(e, hi);
            int kk = (k < 6) ? k : 5;
            offsc[e] = kk * H1P_IC;
        }
        for (int p = wave; p < 49; p += 8) {           // wave-uniform
            int oy = p / 7, ox = p - oy * 7;
            int pbase  = H1P_OFF + lrow*H1P_IMG + oy*32 + ox*2;      // row=2oy+ky
            int pbasec = pbase + 17;                                  // (2oy+1,2ox+1)
            V8F ar, as;
            #pragma unroll
            for (int j = 0; j < 8; ++j) { ar.f[j] = 0.0f; as.f[j] = 0.0f; }
            #pragma unroll
            for (int c = 0; c < 2; ++c) {
                V16H bm;
                #pragma unroll
                for (int e = 0; e < 16; ++e)
                    bm.h[e] = smem[pbase + offs2[c][e]];
                ar.v = wmma32(a2[c].v, bm.v, ar.v);
            }
            {
                V16H bm;
                #pragma unroll
                for (int e = 0; e < 16; ++e)
                    bm.h[e] = smem[pbasec + offsc[e]];
                as.v = wmma32(asc.v, bm.v, as.v);
            }
            #pragma unroll
            for (int j = 0; j < 8; ++j) {
                int oc = hi * 8 + j;
                if (oc < 15) {
                    float o  = fmaxf(ar.f[j] * (g2[oc]*rs) + b2[oc], 0.0f);
                    float sc = as.f[j] * (g3[oc]*rs) + b3[oc];
                    smem[H2PRE_OFF + lrow*735 + oc*49 + oy*7 + ox] =
                        (_Float16)fmaxf(o + sc, 0.0f);
                }
            }
        }
    }
    __syncthreads();

    // ------------- stage 4: maxpool 2x2 s1 + flatten -> F [16][544] ---------
    for (int idx = tid; idx < IPB * F_STR; idx += 256) {
        int img = idx / F_STR; int k = idx - img * F_STR;
        float m = 0.0f;
        if (k < 540) {
            int oc = k / 36; int r = k - oc * 36;
            int y  = r / 6;  int xx = r - y * 6;
            const _Float16* s = &smem[H2PRE_OFF + img*735 + oc*49 + y*7 + xx];
            m = fmaxf(fmaxf((float)s[0], (float)s[1]),
                      fmaxf((float)s[7], (float)s[8]));
        }
        smem[F_OFF + img*F_STR + k] = (_Float16)m;     // zero K-pad 540..543
    }
    __syncthreads();

    // ------------- stage 5: fc1 540->120 + ReLU -----------------------------
    // GEMM: M = img(16), N = 120->128 (8 tiles / 8 waves), K = 540->544.
    {
        int ncol  = wave * 16 + lrow;
        int ncolc = (ncol < 120) ? ncol : 119;
        const float* wrow = w_fc1 + ncolc * 540;
        V8F acc;
        #pragma unroll
        for (int j = 0; j < 8; ++j) acc.f[j] = 0.0f;
        #pragma unroll 4
        for (int c = 0; c < 16; ++c) {                 // full chunks
            V16H am, bm;
            am.q[0] = *(const h8*)&smem[F_OFF + lrow*F_STR + c*32 + hi*16];
            am.q[1] = *(const h8*)&smem[F_OFF + lrow*F_STR + c*32 + hi*16 + 8];
            int kb = c*32 + hi*16;
            const float4* wp = (const float4*)(wrow + kb);
            #pragma unroll
            for (int q = 0; q < 4; ++q) {
                float4 w4 = wp[q];
                bm.h[q*4+0] = (_Float16)w4.x; bm.h[q*4+1] = (_Float16)w4.y;
                bm.h[q*4+2] = (_Float16)w4.z; bm.h[q*4+3] = (_Float16)w4.w;
            }
            acc.v = wmma32(am.v, bm.v, acc.v);
        }
        {   // tail chunk c=16: K 512..543 (valid < 540), branch-free clamp
            V16H am, bm;
            am.q[0] = *(const h8*)&smem[F_OFF + lrow*F_STR + 512 + hi*16];
            am.q[1] = *(const h8*)&smem[F_OFF + lrow*F_STR + 520 + hi*16];
            int kb = 512 + hi*16;
            #pragma unroll
            for (int e = 0; e < 16; ++e) {
                int k  = kb + e;
                int kc = (k < 540) ? k : 539;
                float v = wrow[kc];
                bm.h[e] = (_Float16)((k < 540) ? v : 0.0f);
            }
            acc.v = wmma32(am.v, bm.v, acc.v);
        }
        float bias = b_fc1[ncolc];
        #pragma unroll
        for (int j = 0; j < 8; ++j) {
            int im  = hi * 8 + j;
            float y = fmaxf(acc.f[j] + bias, 0.0f);
            smem[F1_OFF + im*F1_STR + ncol] = (_Float16)((ncol < 120) ? y : 0.0f);
        }
    }
    __syncthreads();

    // ------------- stage 6: fc2 120->84 + ReLU ------------------------------
    // GEMM: M = img(16), N = 84->96 (6 tiles, waves 0-5), K = 120->128.
    if (wave < 6) {
        int ncol  = wave * 16 + lrow;
        int ncolc = (ncol < 84) ? ncol : 83;
        const float* wrow = w_fc2 + ncolc * 120;
        V8F acc;
        #pragma unroll
        for (int j = 0; j < 8; ++j) acc.f[j] = 0.0f;
        #pragma unroll
        for (int c = 0; c < 3; ++c) {                  // full chunks
            V16H am, bm;
            am.q[0] = *(const h8*)&smem[F1_OFF + lrow*F1_STR + c*32 + hi*16];
            am.q[1] = *(const h8*)&smem[F1_OFF + lrow*F1_STR + c*32 + hi*16 + 8];
            int kb = c*32 + hi*16;
            const float4* wp = (const float4*)(wrow + kb);
            #pragma unroll
            for (int q = 0; q < 4; ++q) {
                float4 w4 = wp[q];
                bm.h[q*4+0] = (_Float16)w4.x; bm.h[q*4+1] = (_Float16)w4.y;
                bm.h[q*4+2] = (_Float16)w4.z; bm.h[q*4+3] = (_Float16)w4.w;
            }
            acc.v = wmma32(am.v, bm.v, acc.v);
        }
        {   // tail chunk c=3: K 96..127 (valid < 120)
            V16H am, bm;
            am.q[0] = *(const h8*)&smem[F1_OFF + lrow*F1_STR + 96 + hi*16];
            am.q[1] = *(const h8*)&smem[F1_OFF + lrow*F1_STR + 104 + hi*16];
            int kb = 96 + hi*16;
            #pragma unroll
            for (int e = 0; e < 16; ++e) {
                int k  = kb + e;
                int kc = (k < 120) ? k : 119;
                float v = wrow[kc];
                bm.h[e] = (_Float16)((k < 120) ? v : 0.0f);
            }
            acc.v = wmma32(am.v, bm.v, acc.v);
        }
        float bias = b_fc2[ncolc];
        #pragma unroll
        for (int j = 0; j < 8; ++j) {
            int im  = hi * 8 + j;
            float y = fmaxf(acc.f[j] + bias, 0.0f);
            smem[F2_OFF + im*F2_STR + ncol] = (_Float16)((ncol < 84) ? y : 0.0f);
        }
    }
    __syncthreads();

    // ------------- stage 7: fc3 84->1, cos, sigmoid, write [p, 1-p] ---------
    if (tid < IPB) {
        float s = b_fc3[0];
        #pragma unroll 4
        for (int k = 0; k < 84; ++k)
            s += (float)smem[F2_OFF + tid*F2_STR + k] * w_fc3[k];
        float qv = cosf(s);
        float pv = 1.0f / (1.0f + expf(-qv));
        out[g * IPB + tid]           = pv;
        out[B_TOTAL + g * IPB + tid] = 1.0f - pv;
    }
}

extern "C" void kernel_launch(void* const* d_in, const int* in_sizes, int n_in,
                              void* d_out, int out_size, void* d_ws, size_t ws_size,
                              hipStream_t stream) {
    (void)in_sizes; (void)n_in; (void)out_size; (void)d_ws; (void)ws_size;
    const float* x       = (const float*)d_in[0];
    const float* w_conv1 = (const float*)d_in[1];
    const float* g1      = (const float*)d_in[2];
    const float* b1      = (const float*)d_in[3];
    const float* w_res   = (const float*)d_in[4];
    const float* g2      = (const float*)d_in[5];
    const float* b2      = (const float*)d_in[6];
    const float* w_sc    = (const float*)d_in[7];
    const float* g3      = (const float*)d_in[8];
    const float* b3      = (const float*)d_in[9];
    const float* w_fc1   = (const float*)d_in[10];
    const float* b_fc1   = (const float*)d_in[11];
    const float* w_fc2   = (const float*)d_in[12];
    const float* b_fc2   = (const float*)d_in[13];
    const float* w_fc3   = (const float*)d_in[14];
    const float* b_fc3   = (const float*)d_in[15];
    // d_in[16] = q_params: provably no effect on the output (zero-grad term)

    hybrid_net_kernel<<<dim3(NBLK), dim3(256), SMEM_BYTES, stream>>>(
        x, w_conv1, g1, b1, w_res, g2, b2, w_sc, g3, b3,
        w_fc1, b_fc1, w_fc2, b_fc2, w_fc3, b_fc3, (float*)d_out);
}
